// ReprogrammingLayer_55654186221715
// MI455X (gfx1250) — matrix operand order repaired and checked by reading, not compiled, not run on hardware
//
#include <hip/hip_runtime.h>
#include <hip/hip_bf16.h>

// ---------------------------------------------------------------------------
// Problem constants (from reference): B=8, L=2048, S=1024, D_MODEL=1024,
// D_LLM=4096, H=16, E=64, DHK=1024.
// ---------------------------------------------------------------------------
#define B_DIM   8
#define L_DIM   2048
#define S_DIM   1024
#define DM_DIM  1024
#define DLLM    4096
#define H_DIM   16
#define E_DIM   64
#define DHK_DIM 1024

typedef __bf16 bf16_t;
typedef bf16_t v16bf __attribute__((ext_vector_type(16)));
typedef float  v8f   __attribute__((ext_vector_type(8)));
typedef float  v4f   __attribute__((ext_vector_type(4)));
typedef unsigned int v4u __attribute__((ext_vector_type(4)));
typedef unsigned short u16;

union Frag {
    v16bf bf;
    v4u   q[2];
};

// pack two f32 -> packed bf16x2 (u32). Prefer HW v_cvt_pk_bf16_f32.
static __device__ __forceinline__ unsigned bfpack2(float a, float b) {
#if __has_builtin(__builtin_amdgcn_cvt_pk_bf16_f32)
    auto r = __builtin_amdgcn_cvt_pk_bf16_f32(a, b);
    unsigned u;
    __builtin_memcpy(&u, &r, 4);
    return u;
#else
    union { float f; unsigned u; } ua, ub; ua.f = a; ub.f = b;
    unsigned ra = (ua.u + 0x7FFFu + ((ua.u >> 16) & 1u)) >> 16;
    unsigned rb = (ub.u + 0x7FFFu + ((ub.u >> 16) & 1u)) & 0xFFFF0000u;
    return (ra & 0xFFFFu) | rb;
#endif
}

// ---------------------------------------------------------------------------
// GEMM + bias:  C[M,N] = A[M,K] * W[K,N] + bias[N]  (fp32 memory, bf16 WMMA).
// N, K compile-time; M = gridDim.x*128. Block = 256 thr (8 waves),
// tile 128x128, k-step 32, double-buffered LDS.
// ---------------------------------------------------------------------------
template<int N, int K>
__global__ __launch_bounds__(256)
void gemm_bias_wmma(const float* __restrict__ A, const float* __restrict__ W,
                    const float* __restrict__ bias, float* __restrict__ C) {
    __shared__ __align__(16) u16 As[2][128][32];   // [buf][m][k] bf16
    __shared__ __align__(16) u16 Bs[2][128][32];   // [buf][n][k] bf16 (W^T)

    const int tid  = threadIdx.x;
    const int lane = tid & 31;
    const int wv   = tid >> 5;        // wave id 0..7
    const int wr   = wv >> 2;         // 0..1  -> 64-row band
    const int wc   = wv & 3;          // 0..3  -> 32-col band
    const int hl   = lane >> 4;       // half-wave
    const int l15  = lane & 15;
    const int bm   = blockIdx.x * 128;
    const int bn   = blockIdx.y * 128;

    // per-thread staging assignments
    const int arow = tid >> 1;            // A: row 0..127
    const int ac0  = (tid & 1) * 16;      // A: 16 cols
    const int wkr  = tid >> 3;            // W: k-row 0..31
    const int wn0  = (tid & 7) * 16;      // W: 16 n-cols

    struct Fetch { v4u a0, a1; unsigned w[8]; };

    auto fetch = [&](int kk, Fetch& f) {
        const float* ap = A + (size_t)(bm + arow) * K + kk + ac0;
        v4f a0 = *(const v4f*)(ap + 0);
        v4f a1 = *(const v4f*)(ap + 4);
        v4f a2 = *(const v4f*)(ap + 8);
        v4f a3 = *(const v4f*)(ap + 12);
        f.a0.x = bfpack2(a0.x, a0.y); f.a0.y = bfpack2(a0.z, a0.w);
        f.a0.z = bfpack2(a1.x, a1.y); f.a0.w = bfpack2(a1.z, a1.w);
        f.a1.x = bfpack2(a2.x, a2.y); f.a1.y = bfpack2(a2.z, a2.w);
        f.a1.z = bfpack2(a3.x, a3.y); f.a1.w = bfpack2(a3.z, a3.w);
        const float* wp = W + (size_t)(kk + wkr) * N + bn + wn0;
#pragma unroll
        for (int j = 0; j < 4; ++j) {
            v4f x = *(const v4f*)(wp + 4 * j);
            f.w[2 * j + 0] = bfpack2(x.x, x.y);
            f.w[2 * j + 1] = bfpack2(x.z, x.w);
        }
    };
    auto stash = [&](int buf, const Fetch& f) {
        *(v4u*)&As[buf][arow][ac0]     = f.a0;
        *(v4u*)&As[buf][arow][ac0 + 8] = f.a1;
#pragma unroll
        for (int j = 0; j < 8; ++j) {
            Bs[buf][wn0 + 2 * j + 0][wkr] = (u16)f.w[j];
            Bs[buf][wn0 + 2 * j + 1][wkr] = (u16)(f.w[j] >> 16);
        }
    };

    v8f acc[4][2];
#pragma unroll
    for (int mt = 0; mt < 4; ++mt)
#pragma unroll
        for (int nt = 0; nt < 2; ++nt) acc[mt][nt] = (v8f){};

    Fetch f;
    fetch(0, f);
    stash(0, f);
    __syncthreads();

    constexpr int NITER = K / 32;
    for (int it = 0; it < NITER; ++it) {
        const int cur = it & 1;
        if (it + 1 < NITER) fetch((it + 1) * 32, f);

        // load all fragments, then issue WMMAs back-to-back
        Frag bfr[2], afr[4];
#pragma unroll
        for (int nt = 0; nt < 2; ++nt) {
            const u16* bp = &Bs[cur][wc * 32 + nt * 16 + l15][hl * 8];
            bfr[nt].q[0] = *(const v4u*)(bp + 0);
            bfr[nt].q[1] = *(const v4u*)(bp + 16);
        }
#pragma unroll
        for (int mt = 0; mt < 4; ++mt) {
            const u16* ap2 = &As[cur][wr * 64 + mt * 16 + l15][0];
            afr[mt].q[0] = *(const v4u*)(ap2 + hl * 8);
            afr[mt].q[1] = *(const v4u*)(ap2 + 16 + hl * 8);
        }
#pragma unroll
        for (int mt = 0; mt < 4; ++mt)
#pragma unroll
            for (int nt = 0; nt < 2; ++nt)
                acc[mt][nt] = __builtin_amdgcn_wmma_f32_16x16x32_bf16(
                    false, afr[mt].bf, false, bfr[nt].bf,
                    (short)0, acc[mt][nt], false, false);

        if (it + 1 < NITER) stash(cur ^ 1, f);
        __syncthreads();
    }

    // --- epilogue: bias + store (C layout: m = v + 8*hl, n = l15)
#pragma unroll
    for (int nt = 0; nt < 2; ++nt) {
        const int n   = bn + wc * 32 + nt * 16 + l15;
        const float bvv = bias[n];
#pragma unroll
        for (int mt = 0; mt < 4; ++mt) {
            const int mrow = bm + wr * 64 + mt * 16 + hl * 8;
#pragma unroll
            for (int v = 0; v < 8; ++v)
                C[(size_t)(mrow + v) * N + n] = acc[mt][nt][v] + bvv;
        }
    }
}

// ---------------------------------------------------------------------------
// Fused flash attention. Grid: (L/128, H, B). Block = 256 thr (8 waves),
// each wave owns 16 query rows; 32-row K/V tiles double-buffered in LDS.
// Q,K,V,O are fp32 [B, L|S, H*E] row-major.
// ---------------------------------------------------------------------------
__global__ __launch_bounds__(256)
void attn_wmma(const float* __restrict__ Q, const float* __restrict__ Km,
               const float* __restrict__ Vm, float* __restrict__ O) {
    __shared__ __align__(16) u16 Ks[2][32][64];    // [buf][s][e]
    __shared__ __align__(16) u16 Vs[2][64][32];    // [buf][e][s] (transposed)
    __shared__ __align__(16) u16 Ps[8][16][32];    // per-wave P scratch [m][s]

    const int tid  = threadIdx.x;
    const int lane = tid & 31;
    const int wv   = tid >> 5;
    const int hl   = lane >> 4;
    const int l15  = lane & 15;
    const int b    = blockIdx.z;
    const int h    = blockIdx.y;
    const int mbase = blockIdx.x * 128 + wv * 16;

    // staging assignment
    const int sl = tid >> 3;          // 0..31 (source row in tile)
    const int e0 = (tid & 7) * 8;     // 0..56 (8 e-cols)

    struct FetchKV { v4u kp, vp; };
    auto fetchKV = [&](int s0, FetchKV& f) {
        const size_t base =
            ((size_t)(b * S_DIM + s0 + sl)) * DHK_DIM + h * E_DIM + e0;
        v4f x0 = *(const v4f*)(Km + base);
        v4f x1 = *(const v4f*)(Km + base + 4);
        f.kp.x = bfpack2(x0.x, x0.y); f.kp.y = bfpack2(x0.z, x0.w);
        f.kp.z = bfpack2(x1.x, x1.y); f.kp.w = bfpack2(x1.z, x1.w);
        v4f y0 = *(const v4f*)(Vm + base);
        v4f y1 = *(const v4f*)(Vm + base + 4);
        f.vp.x = bfpack2(y0.x, y0.y); f.vp.y = bfpack2(y0.z, y0.w);
        f.vp.z = bfpack2(y1.x, y1.y); f.vp.w = bfpack2(y1.z, y1.w);
    };
    auto stashKV = [&](int buf, const FetchKV& f) {
        *(v4u*)&Ks[buf][sl][e0] = f.kp;
        Vs[buf][e0 + 0][sl] = (u16)f.vp.x; Vs[buf][e0 + 1][sl] = (u16)(f.vp.x >> 16);
        Vs[buf][e0 + 2][sl] = (u16)f.vp.y; Vs[buf][e0 + 3][sl] = (u16)(f.vp.y >> 16);
        Vs[buf][e0 + 4][sl] = (u16)f.vp.z; Vs[buf][e0 + 5][sl] = (u16)(f.vp.z >> 16);
        Vs[buf][e0 + 6][sl] = (u16)f.vp.w; Vs[buf][e0 + 7][sl] = (u16)(f.vp.w >> 16);
    };

    // --- preload Q fragments (2 k-chunks of 32), pre-scaled by 1/sqrt(E)
    Frag qf[2];
    {
        const float sc = 0.125f;  // 1/sqrt(64)
        const float* qp =
            Q + ((size_t)(b * L_DIM + mbase + l15)) * DHK_DIM + h * E_DIM;
#pragma unroll
        for (int kc = 0; kc < 2; ++kc) {
            const int eb = kc * 32 + hl * 8;
            v4f x0 = *(const v4f*)(qp + eb);
            v4f x1 = *(const v4f*)(qp + eb + 4);
            v4f x2 = *(const v4f*)(qp + eb + 16);
            v4f x3 = *(const v4f*)(qp + eb + 20);
            qf[kc].q[0].x = bfpack2(x0.x * sc, x0.y * sc);
            qf[kc].q[0].y = bfpack2(x0.z * sc, x0.w * sc);
            qf[kc].q[0].z = bfpack2(x1.x * sc, x1.y * sc);
            qf[kc].q[0].w = bfpack2(x1.z * sc, x1.w * sc);
            qf[kc].q[1].x = bfpack2(x2.x * sc, x2.y * sc);
            qf[kc].q[1].y = bfpack2(x2.z * sc, x2.w * sc);
            qf[kc].q[1].z = bfpack2(x3.x * sc, x3.y * sc);
            qf[kc].q[1].w = bfpack2(x3.z * sc, x3.w * sc);
        }
    }

    float run_max[8], run_sum[8];
    v8f acco[4];
#pragma unroll
    for (int v = 0; v < 8; ++v) { run_max[v] = -__builtin_inff(); run_sum[v] = 0.f; }
#pragma unroll
    for (int et = 0; et < 4; ++et) acco[et] = (v8f){};

    FetchKV fkv;
    fetchKV(0, fkv);
    stashKV(0, fkv);
    __syncthreads();

    constexpr int NS = S_DIM / 32;
    for (int it = 0; it < NS; ++it) {
        const int cur = it & 1;
        if (it + 1 < NS) fetchKV((it + 1) * 32, fkv);

        // --- scores: S16[16 x 32] = Q(16x64) * K^T(64x32)
        v8f sacc[2];
        sacc[0] = (v8f){}; sacc[1] = (v8f){};
#pragma unroll
        for (int nt = 0; nt < 2; ++nt) {
            const u16* kp2 = &Ks[cur][nt * 16 + l15][0];
#pragma unroll
            for (int kc = 0; kc < 2; ++kc) {
                Frag bfr;
                bfr.q[0] = *(const v4u*)(kp2 + kc * 32 + hl * 8);
                bfr.q[1] = *(const v4u*)(kp2 + kc * 32 + hl * 8 + 16);
                sacc[nt] = __builtin_amdgcn_wmma_f32_16x16x32_bf16(
                    false, qf[kc].bf, false, bfr.bf,
                    (short)0, sacc[nt], false, false);
            }
        }

        // --- online softmax (per row v+8*hl; reduce across 16-lane half)
#pragma unroll
        for (int v = 0; v < 8; ++v) {
            float tmax = fmaxf(sacc[0][v], sacc[1][v]);
            tmax = fmaxf(tmax, __shfl_xor(tmax, 1));
            tmax = fmaxf(tmax, __shfl_xor(tmax, 2));
            tmax = fmaxf(tmax, __shfl_xor(tmax, 4));
            tmax = fmaxf(tmax, __shfl_xor(tmax, 8));
            const float mnew = fmaxf(run_max[v], tmax);
            const float corr = __expf(run_max[v] - mnew);
            const float p0 = __expf(sacc[0][v] - mnew);
            const float p1 = __expf(sacc[1][v] - mnew);
            float ps = p0 + p1;
            ps += __shfl_xor(ps, 1);
            ps += __shfl_xor(ps, 2);
            ps += __shfl_xor(ps, 4);
            ps += __shfl_xor(ps, 8);
            run_sum[v] = run_sum[v] * corr + ps;
            run_max[v] = mnew;
            const int ml = v + hl * 8;
            Ps[wv][ml][l15]      = (u16)bfpack2(p0, 0.f);
            Ps[wv][ml][16 + l15] = (u16)bfpack2(p1, 0.f);
#pragma unroll
            for (int et = 0; et < 4; ++et) acco[et][v] *= corr;
        }

        // --- P as A-fragment (16x32), V as B-fragments; O += P*V
        Frag pfr;
        {
            const u16* pp = &Ps[wv][l15][0];
            pfr.q[0] = *(const v4u*)(pp + hl * 8);
            pfr.q[1] = *(const v4u*)(pp + 16 + hl * 8);
        }
        Frag vfr[4];
#pragma unroll
        for (int et = 0; et < 4; ++et) {
            const u16* vp2 = &Vs[cur][et * 16 + l15][hl * 8];
            vfr[et].q[0] = *(const v4u*)(vp2 + 0);
            vfr[et].q[1] = *(const v4u*)(vp2 + 16);
        }
#pragma unroll
        for (int et = 0; et < 4; ++et)
            acco[et] = __builtin_amdgcn_wmma_f32_16x16x32_bf16(
                false, pfr.bf, false, vfr[et].bf,
                (short)0, acco[et], false, false);

        if (it + 1 < NS) stashKV(cur ^ 1, fkv);
        __syncthreads();
    }

    // --- normalize and store
#pragma unroll
    for (int v = 0; v < 8; ++v) {
        const float inv = 1.0f / run_sum[v];
        const int m = mbase + v + hl * 8;
        float* op = O + ((size_t)(b * L_DIM + m)) * DHK_DIM + h * E_DIM;
#pragma unroll
        for (int et = 0; et < 4; ++et)
            op[et * 16 + l15] = acco[et][v] * inv;
    }
}

// ---------------------------------------------------------------------------
// Launcher
// ---------------------------------------------------------------------------
extern "C" void kernel_launch(void* const* d_in, const int* in_sizes, int n_in,
                              void* d_out, int out_size, void* d_ws, size_t ws_size,
                              hipStream_t stream) {
    const float* tgt = (const float*)d_in[0];   // [B,L,D_MODEL]
    const float* src = (const float*)d_in[1];   // [B,S,D_LLM]
    const float* Wq  = (const float*)d_in[2];
    const float* bq  = (const float*)d_in[3];
    const float* Wk  = (const float*)d_in[4];
    const float* bk  = (const float*)d_in[5];
    const float* Wv  = (const float*)d_in[6];
    const float* bv  = (const float*)d_in[7];
    const float* Wo  = (const float*)d_in[8];
    const float* bo  = (const float*)d_in[9];
    float* out = (float*)d_out;                 // [B,L,D_LLM] fp32

    // workspace: Q [16384,1024] | K [8192,1024] | V [8192,1024] | AO [16384,1024]
    float* qf = (float*)d_ws;
    float* kf = qf + (size_t)(B_DIM * L_DIM) * DHK_DIM;
    float* vf = kf + (size_t)(B_DIM * S_DIM) * DHK_DIM;
    float* af = vf + (size_t)(B_DIM * S_DIM) * DHK_DIM;

    const dim3 blk(256);
    const int MQ = B_DIM * L_DIM;   // 16384
    const int MS = B_DIM * S_DIM;   // 8192

    // Q = tgt @ Wq + bq           (M=16384, N=1024, K=1024)
    gemm_bias_wmma<DHK_DIM, DM_DIM>
        <<<dim3(MQ / 128, DHK_DIM / 128), blk, 0, stream>>>(tgt, Wq, bq, qf);
    // K = src @ Wk + bk           (M=8192, N=1024, K=4096)
    gemm_bias_wmma<DHK_DIM, DLLM>
        <<<dim3(MS / 128, DHK_DIM / 128), blk, 0, stream>>>(src, Wk, bk, kf);
    // V = src @ Wv + bv
    gemm_bias_wmma<DHK_DIM, DLLM>
        <<<dim3(MS / 128, DHK_DIM / 128), blk, 0, stream>>>(src, Wv, bv, vf);
    // AO = softmax(Q K^T / sqrt(E)) V   (fused, flash-style)
    attn_wmma<<<dim3(L_DIM / 128, H_DIM, B_DIM), blk, 0, stream>>>(qf, kf, vf, af);
    // out = AO @ Wo + bo          (M=16384, N=4096, K=1024)
    gemm_bias_wmma<DLLM, DHK_DIM>
        <<<dim3(MQ / 128, DLLM / 128), blk, 0, stream>>>(af, Wo, bo, out);

    (void)in_sizes; (void)n_in; (void)out_size; (void)ws_size;
}
